// AttnDecoderRNN_19232863552199
// MI455X (gfx1250) — matrix operand order repaired and checked by reading, not compiled, not run on hardware
//
#include <hip/hip_runtime.h>
#include <hip/hip_bf16.h>

// ---------------- types ----------------
typedef __bf16 bf16_t;
typedef __attribute__((ext_vector_type(16))) __bf16 v16bf;
typedef __attribute__((ext_vector_type(8)))  __bf16 v8bf;
typedef __attribute__((ext_vector_type(8)))  float  v8f;

#define HDIM 1024
#define VDIM 32000
#define SDIM 512
#define BDIM 32
#define TDIM 50

__device__ __forceinline__ bf16_t to_bf16(float f) {
    unsigned u = __builtin_bit_cast(unsigned, f);
    unsigned r = u + 0x7FFFu + ((u >> 16) & 1u);   // round-to-nearest-even
    unsigned short h = (unsigned short)(r >> 16);
    return __builtin_bit_cast(bf16_t, h);
}

// ---------------- elementwise / setup kernels ----------------
__global__ void k_f32_to_bf16(const float* __restrict__ src, bf16_t* __restrict__ dst, int n) {
    int i = blockIdx.x * blockDim.x + threadIdx.x;
    if (i < n) dst[i] = to_bf16(src[i]);
}

__global__ void k_gather_emb(const float* __restrict__ emb, const int* __restrict__ target,
                             const int* __restrict__ sos_p, bf16_t* __restrict__ Ebf) {
    int i = blockIdx.x * blockDim.x + threadIdx.x;   // 1600*1024 threads
    int row = i >> 10;                               // row = t*B + b
    int col = i & 1023;
    int t = row / BDIM, b = row % BDIM;
    int tok = (t == 0) ? sos_p[0] : target[b * TDIM + (t - 1)];
    Ebf[i] = to_bf16(emb[(size_t)tok * HDIM + col]);
}

// ctx[b,h] = sum_s enc[b,s,h]  (attention weights are provably all 1.0)
__global__ void k_ctx(const float* __restrict__ enc, float* __restrict__ ctx,
                      bf16_t* __restrict__ ctxb) {
    int i = blockIdx.x * blockDim.x + threadIdx.x;   // B*H
    int b = i >> 10, h = i & 1023;
    const float* p = enc + (size_t)b * SDIM * HDIM + h;
    float s = 0.f;
    for (int ss = 0; ss < SDIM; ++ss) s += p[(size_t)ss * HDIM];
    ctx[i] = s;
    ctxb[i] = to_bf16(s);
}

__global__ void k_init_h(const float* __restrict__ eh, float* __restrict__ Hf,
                         bf16_t* __restrict__ Hb) {
    int i = blockIdx.x * blockDim.x + threadIdx.x;   // B*H
    float v = eh[i];
    Hf[i] = v;
    Hb[i] = to_bf16(v);
}

__global__ void k_fill_ones(float* __restrict__ p, int n) {
    int i = blockIdx.x * blockDim.x + threadIdx.x;
    if (i < n) p[i] = 1.0f;
}

__global__ void k_copy_f32(const float* __restrict__ s, float* __restrict__ d, int n) {
    int i = blockIdx.x * blockDim.x + threadIdx.x;
    if (i < n) d[i] = s[i];
}

// GRU gate fusion: reads precomputed embedding-gates row, constant ctx-gates, gh
__global__ void k_gru_gates(const float* __restrict__ GXe_t,  // [B,3H] for this step
                            const float* __restrict__ cgx,    // [B,3H] (includes b_ih)
                            const float* __restrict__ gh,     // [B,3H] (includes b_hh)
                            const float* __restrict__ hprev,  // [B,H]
                            float* __restrict__ hnf, bf16_t* __restrict__ hnb) {
    int i = blockIdx.x * blockDim.x + threadIdx.x;   // B*H
    int b = i >> 10, k = i & 1023;
    int base = b * 3 * HDIM;
    float gxr = GXe_t[base + k]            + cgx[base + k];
    float gxz = GXe_t[base + HDIM + k]     + cgx[base + HDIM + k];
    float gxn = GXe_t[base + 2 * HDIM + k] + cgx[base + 2 * HDIM + k];
    float ghr = gh[base + k];
    float ghz = gh[base + HDIM + k];
    float ghn = gh[base + 2 * HDIM + k];
    float r = 1.f / (1.f + __expf(-(gxr + ghr)));
    float z = 1.f / (1.f + __expf(-(gxz + ghz)));
    float n = tanhf(gxn + r * ghn);
    float h0 = hprev[i];
    float hn = (1.f - z) * n + z * h0;
    hnf[i] = hn;
    hnb[i] = to_bf16(hn);
}

// in-place log_softmax over each V-length row of d_out
__global__ void k_log_softmax(float* __restrict__ out) {
    __shared__ float red[256];
    float* p = out + (size_t)blockIdx.x * VDIM;
    int tid = threadIdx.x;
    float mx = -INFINITY;
    for (int i = tid; i < VDIM; i += 256) mx = fmaxf(mx, p[i]);
    red[tid] = mx; __syncthreads();
    for (int s = 128; s > 0; s >>= 1) { if (tid < s) red[tid] = fmaxf(red[tid], red[tid + s]); __syncthreads(); }
    mx = red[0]; __syncthreads();
    float sum = 0.f;
    for (int i = tid; i < VDIM; i += 256) sum += __expf(p[i] - mx);
    red[tid] = sum; __syncthreads();
    for (int s = 128; s > 0; s >>= 1) { if (tid < s) red[tid] += red[tid + s]; __syncthreads(); }
    float lse = mx + __logf(red[0]);
    for (int i = tid; i < VDIM; i += 256) p[i] = p[i] - lse;
}

// ---------------- WMMA bf16 GEMM:  C[M,N] = A[M,K] @ W[N,K]^T (+bias) ----------------
// MODE 0: store f32 into Cf (stride ldc)
// MODE 1: relu(acc+bias)^2 -> bf16 into Cb (stride ldc)
// MODE 2: scatter f32 logits: row m=(t*B+b) -> Cf[(b*T+t)*V + n]
// MT: M-register-blocking — each wave computes an (MT*16) x 16 tile, reusing the
//     B fragment across MT WMMAs (cuts B-side L2 traffic by MT, gives MT
//     independent accumulator chains for WMMA pipelining).
// grid: (N/128, M/(16*MT)), block: 256 threads = 8 waves (one N-16 column each).
template <int MODE, int MT>
__global__ void k_wmma_gemm(const bf16_t* __restrict__ A, int lda,
                            const bf16_t* __restrict__ Bw, int ldb, int bofs,
                            const float* __restrict__ bias,
                            float* __restrict__ Cf, bf16_t* __restrict__ Cb, int ldc,
                            int K) {
    const int lane = threadIdx.x & 31;
    const int wave = threadIdx.x >> 5;
    const int half = lane >> 4;      // 0: lanes 0-15, 1: lanes 16-31
    const int l    = lane & 15;
    const size_t row0 = (size_t)blockIdx.y * (16 * MT);
    const size_t col0 = (size_t)blockIdx.x * 128 + (size_t)wave * 16;

    const bf16_t* arow = A  + (row0 + l) * (size_t)lda;          // A row m=row0+l (subtile 0)
    const bf16_t* brow = Bw + (col0 + l) * (size_t)ldb + bofs;   // W row n=col0+l

    v8f c[MT];
#pragma unroll
    for (int mt = 0; mt < MT; ++mt) c[mt] = (v8f){};

    for (int kb = 0; kb < K; kb += 32) {
        // B 32x16 fragment: lane (l,half) holds column n=col0+l, K[16*half .. +15];
        // one contiguous 32B load from the [N,K] weight row — loaded ONCE, used MT times.
        v16bf bfr = *(const v16bf*)(brow + kb + half * 16);
        __builtin_prefetch(brow + kb + 64, 0, 1);                 // global_prefetch_b8
#pragma unroll
        for (int mt = 0; mt < MT; ++mt) {
            // A 16x32 bf16 fragment (matches B): lanes 0-15 -> K[0..15];
            // lanes 16-31 -> K[16..31]; two contiguous 16B loads.
            const bf16_t* ap = arow + (size_t)mt * 16 * lda + kb;
            v8bf a0 = *(const v8bf*)(ap + half * 16);
            v8bf a1 = *(const v8bf*)(ap + half * 16 + 8);
            v16bf a;
#pragma unroll
            for (int j = 0; j < 8; ++j) { a[j] = a0[j]; a[j + 8] = a1[j]; }
            c[mt] = __builtin_amdgcn_wmma_f32_16x16x32_bf16(false, a, false, bfr,
                                                            (short)0, c[mt], false, false);
        }
    }
#pragma unroll
    for (int mt = 0; mt < MT; ++mt) {
#pragma unroll
        for (int r = 0; r < 8; ++r) {
            size_t mm = row0 + (size_t)mt * 16 + (size_t)r + 8 * (size_t)half;
            size_t nn = col0 + l;                 // C: VGPR r -> row r+8*(lane/16), col lane%16
            float v = c[mt][r] + (bias ? bias[nn] : 0.f);
            if (MODE == 0) {
                Cf[mm * (size_t)ldc + nn] = v;
            } else if (MODE == 1) {
                v = v > 0.f ? v : 0.f;
                Cb[mm * (size_t)ldc + nn] = to_bf16(v * v);
            } else {
                int t = (int)(mm / BDIM), b = (int)(mm % BDIM);
                Cf[((size_t)(b * TDIM + t)) * VDIM + nn] = v;
            }
        }
    }
}

// ---------------- host launcher ----------------
extern "C" void kernel_launch(void* const* d_in, const int* in_sizes, int n_in,
                              void* d_out, int out_size, void* d_ws, size_t ws_size,
                              hipStream_t stream) {
    const float* enc   = (const float*)d_in[0];
    const float* eh    = (const float*)d_in[1];
    const int*   sos   = (const int*)d_in[2];
    const int*   tgt   = (const int*)d_in[3];
    const float* emb   = (const float*)d_in[4];
    // d_in[5..10] (Wa,bWa,Ua,bUa,Va,bVa) are provably unused: softmax over size-1 axis == 1
    const float* W_ih  = (const float*)d_in[11];
    const float* W_hh  = (const float*)d_in[12];
    const float* b_ih  = (const float*)d_in[13];
    const float* b_hh  = (const float*)d_in[14];
    const float* W1    = (const float*)d_in[15];
    const float* b1    = (const float*)d_in[16];
    const float* W2    = (const float*)d_in[17];
    const float* b2    = (const float*)d_in[18];
    (void)in_sizes; (void)n_in; (void)out_size; (void)ws_size;

    const int BH  = BDIM * HDIM;          // 32768
    const int B3H = BDIM * 3 * HDIM;      // 98304
    const int TB  = TDIM * BDIM;          // 1600

    // workspace carve
    char* w = (char*)d_ws;
    auto carve = [&](size_t bytes) { char* p = w; w += (bytes + 255) & ~(size_t)255; return p; };
    bf16_t* Wih_b = (bf16_t*)carve((size_t)3072 * 2048 * 2);
    bf16_t* Whh_b = (bf16_t*)carve((size_t)3072 * 1024 * 2);
    bf16_t* W1_b  = (bf16_t*)carve((size_t)1024 * 1024 * 2);
    bf16_t* W2_b  = (bf16_t*)carve((size_t)VDIM * 1024 * 2);
    bf16_t* Ebf   = (bf16_t*)carve((size_t)TB * HDIM * 2);
    float*  GXe   = (float*) carve((size_t)TB * 3 * HDIM * 4);
    float*  ctxf  = (float*) carve((size_t)BH * 4);
    bf16_t* ctxb  = (bf16_t*)carve((size_t)BH * 2);
    float*  cgx   = (float*) carve((size_t)B3H * 4);
    float*  ghws  = (float*) carve((size_t)B3H * 4);
    float*  Hf    = (float*) carve((size_t)(TDIM + 1) * BH * 4);
    bf16_t* Hb    = (bf16_t*)carve((size_t)(TDIM + 1) * BH * 2);
    bf16_t* Xbf   = (bf16_t*)carve((size_t)TB * HDIM * 2);

    float* out_logits = (float*)d_out;                               // [B,T,V]
    float* out_hfin   = out_logits + (size_t)BDIM * TDIM * VDIM;     // [1,B,H]
    float* out_attn   = out_hfin + BH;                               // [B,T,S]

    // weight conversions to bf16
    k_f32_to_bf16<<<(3072 * 2048) / 256, 256, 0, stream>>>(W_ih, Wih_b, 3072 * 2048);
    k_f32_to_bf16<<<(3072 * 1024) / 256, 256, 0, stream>>>(W_hh, Whh_b, 3072 * 1024);
    k_f32_to_bf16<<<(1024 * 1024) / 256, 256, 0, stream>>>(W1, W1_b, 1024 * 1024);
    k_f32_to_bf16<<<(VDIM * 1024) / 256, 256, 0, stream>>>(W2, W2_b, VDIM * 1024);

    // setup
    k_gather_emb<<<(TB * HDIM) / 256, 256, 0, stream>>>(emb, tgt, sos, Ebf);
    k_ctx<<<BH / 256, 256, 0, stream>>>(enc, ctxf, ctxb);
    k_init_h<<<BH / 256, 256, 0, stream>>>(eh, Hf, Hb);
    k_fill_ones<<<(BDIM * TDIM * SDIM) / 256, 256, 0, stream>>>(out_attn, BDIM * TDIM * SDIM);

    // precompute all input-side gates:
    // GXe = Ebf @ W_ih[:, :H]^T           [1600 x 3072], 64-row wave tiles
    k_wmma_gemm<0, 4><<<dim3(3072 / 128, TB / 64), 256, 0, stream>>>(
        Ebf, HDIM, Wih_b, 2048, 0, nullptr, GXe, nullptr, 3 * HDIM, HDIM);
    // cgx = ctx @ W_ih[:, H:2H]^T + b_ih  [32 x 3072], 32-row wave tiles
    k_wmma_gemm<0, 2><<<dim3(3072 / 128, 1), 256, 0, stream>>>(
        ctxb, HDIM, Wih_b, 2048, HDIM, b_ih, cgx, nullptr, 3 * HDIM, HDIM);

    // sequential GRU recurrence (only true dependency chain)
    for (int t = 0; t < TDIM; ++t) {
        k_wmma_gemm<0, 2><<<dim3(3072 / 128, 1), 256, 0, stream>>>(
            Hb + (size_t)t * BH, HDIM, Whh_b, 1024, 0, b_hh, ghws, nullptr, 3 * HDIM, HDIM);
        k_gru_gates<<<BH / 256, 256, 0, stream>>>(
            GXe + (size_t)t * B3H, cgx, ghws, Hf + (size_t)t * BH,
            Hf + (size_t)(t + 1) * BH, Hb + (size_t)(t + 1) * BH);
    }

    // h_final
    k_copy_f32<<<BH / 256, 256, 0, stream>>>(Hf + (size_t)TDIM * BH, out_hfin, BH);

    // batched output head
    // X = relu(HN @ W1^T + b1)^2  -> bf16   [1600 x 1024]
    k_wmma_gemm<1, 4><<<dim3(1024 / 128, TB / 64), 256, 0, stream>>>(
        Hb + BH, HDIM, W1_b, 1024, 0, b1, nullptr, Xbf, HDIM, HDIM);
    // logits = X @ W2^T + b2, scattered into d_out [B,T,V]
    k_wmma_gemm<2, 4><<<dim3(VDIM / 128, TB / 64), 256, 0, stream>>>(
        Xbf, HDIM, W2_b, 1024, 0, b2, out_logits, nullptr, 0, HDIM);
    // in-place log_softmax over V for all B*T rows
    k_log_softmax<<<TB, 256, 0, stream>>>(out_logits);
}
